// AutoregressiveGRU_11879879541255
// MI455X (gfx1250) — matrix operand order repaired and checked by reading, not compiled, not run on hardware
//
#include <hip/hip_runtime.h>

// ---------------------------------------------------------------------------
// Autoregressive GRU on MI455X (gfx1250) — single persistent kernel.
// B=256, D=1024, T=128.  For t>=1, inp==h, so each step is one GEMM
// h[256x1024] @ [W|U][1024x6144] in bf16 WMMA (fp32 accumulate) + gates.
// Weights converted once to bf16 fragments (12 MB, L2-resident).
// All 128 steps run inside one kernel with a device-scope grid barrier.
// Step 0 (inp==0) is peeled so the steady-state inner loop is branch-free,
// and the k-loop is unrolled x4 to recover load/WMMA software pipelining.
// ---------------------------------------------------------------------------

typedef __attribute__((ext_vector_type(16))) __bf16 v16bf;
typedef __attribute__((ext_vector_type(8)))  float  v8f;

static constexpr int DDIM   = 1024;          // hidden size
static constexpr int B3     = 3 * DDIM;      // 3072 gate columns
static constexpr int NB     = 256;           // batch
static constexpr int TSTEPS = 128;           // decode steps
static constexpr int KT     = DDIM / 32;     // 32 k-tiles of depth 32
static constexpr int NBLK   = 128;           // persistent blocks (x 8 waves = 1024 tiles)
// fragment element counts (ushort elements)
static constexpr size_t WFRAG_ELEMS = (size_t)(B3 / 16) * KT * 32 * 16; // 3,145,728
static constexpr size_t AFRAG_ELEMS = (size_t)(NB / 16) * KT * 32 * 16; //   262,144

__device__ __forceinline__ unsigned short f2bf(float f) {
  union { float f; unsigned u; } c; c.f = f;
  unsigned u = c.u;
  return (unsigned short)((u + 0x7FFFu + ((u >> 16) & 1u)) >> 16); // RNE
}

__device__ __forceinline__ float sigmoid_f(float x) {
  return 1.0f / (1.0f + __expf(-x));
}
__device__ __forceinline__ float tanh_f(float x) {
  return 1.0f - 2.0f / (1.0f + __expf(2.0f * x));   // stable at +/- inf
}

// ---------------------------------------------------------------------------
// Pack a [1024 x 3072] fp32 row-major weight into bf16 WMMA B-fragments.
// 32x16 bf16 B-tile: lane l holds column n = l%16; lanes 0-15 hold K=0..7,16..23
// lanes 16-31 hold K=8..15,24..31.  idx = ((nt*32 + kt)*32 + lane)*16 + j
// ---------------------------------------------------------------------------
__global__ __launch_bounds__(256) void pack_weight_kernel(
    const float* __restrict__ W, unsigned short* __restrict__ frag) {
  int idx  = blockIdx.x * 256 + threadIdx.x;   // < 3,145,728
  int j    = idx & 15;
  int lane = (idx >> 4) & 31;
  int kt   = (idx >> 9) & 31;
  int nt   = idx >> 14;
  int k    = kt * 32 + (j & 7) + ((j >> 3) << 4) + ((lane >= 16) ? 8 : 0);
  int n    = nt * 16 + (lane & 15);
  frag[idx] = f2bf(W[k * B3 + n]);
}

// ---------------------------------------------------------------------------
// Pack h [256 x 1024] fp32 into bf16 WMMA A-fragments (16x32 bf16 A-tile).
// lane<16: row m=lane, K in {0..7 (j=K), 16..23 (j=K-8)};
// lane>=16: row m=lane-16, K in {8..15 (j=K-8), 24..31 (j=K-16)}.
// ---------------------------------------------------------------------------
__global__ __launch_bounds__(256) void pack_a_kernel(
    const float* __restrict__ h, unsigned short* __restrict__ frag) {
  int idx = blockIdx.x * 256 + threadIdx.x;    // < 262,144
  int m   = idx >> 10;
  int k   = idx & 1023;
  int mt = m >> 4, ktile = k >> 5, kin = k & 31;
  int lane = (m & 15) + (((kin >> 3) & 1) << 4);
  int j    = (kin & 7) + ((kin >> 4) << 3);
  frag[(((mt * KT + ktile) * 32 + lane) << 4) + j] = f2bf(h[idx]);
}

__global__ void init_barrier_kernel(unsigned* __restrict__ bar) {
  if (blockIdx.x == 0 && threadIdx.x == 0) { bar[0] = 0u; bar[1] = 0u; }
}

// ---------------------------------------------------------------------------
// Per-wave constants (computed once, reused every step).
// ---------------------------------------------------------------------------
struct TileCtx {
  int lane, nt, mt, half;
  int bz0, br0, bh0;       // v16bf offsets of the three gate columns
  int ncol;                // output column within D
  float bzv, brv, bhv;     // biases for z | r | h
  int ktile, lane2b, j2;   // scatter constants for next-step A fragments
};

// One GRU step for this wave's 16x16 tile.  FIRST=true: inp==0 (skip W GEMM).
template <bool FIRST>
__device__ __forceinline__ void gru_step_body(
    const TileCtx& c,
    const unsigned short* __restrict__ afrag,
    const float* __restrict__ hcur,
    const v16bf* __restrict__ wv, const v16bf* __restrict__ uv,
    float* __restrict__ hnext,
    unsigned short* __restrict__ afrag_next,
    float* __restrict__ out_t) {
  const v16bf* ap = reinterpret_cast<const v16bf*>(afrag) + (c.mt * KT) * 32 + c.lane;

  v8f axz = {}, axr = {}, axh = {}, ahz = {}, ahr = {}, ahh = {};
#pragma unroll 4
  for (int kt = 0; kt < KT; ++kt) {
    v16bf a = ap[kt * 32];
    const int o = kt * 32;
    if (!FIRST) {
      axz = __builtin_amdgcn_wmma_f32_16x16x32_bf16(false, a, false, wv[c.bz0 + o], (short)0, axz, false, false);
      axr = __builtin_amdgcn_wmma_f32_16x16x32_bf16(false, a, false, wv[c.br0 + o], (short)0, axr, false, false);
      axh = __builtin_amdgcn_wmma_f32_16x16x32_bf16(false, a, false, wv[c.bh0 + o], (short)0, axh, false, false);
    }
    ahz = __builtin_amdgcn_wmma_f32_16x16x32_bf16(false, a, false, uv[c.bz0 + o], (short)0, ahz, false, false);
    ahr = __builtin_amdgcn_wmma_f32_16x16x32_bf16(false, a, false, uv[c.br0 + o], (short)0, ahr, false, false);
    ahh = __builtin_amdgcn_wmma_f32_16x16x32_bf16(false, a, false, uv[c.bh0 + o], (short)0, ahh, false, false);
  }

#pragma unroll
  for (int v = 0; v < 8; ++v) {
    int m = (c.mt << 4) + v + (c.half << 3);
    float z  = sigmoid_f((FIRST ? 0.0f : axz[v]) + c.bzv + ahz[v]);
    float r  = sigmoid_f((FIRST ? 0.0f : axr[v]) + c.brv + ahr[v]);
    float hh = tanh_f((FIRST ? 0.0f : axh[v]) + c.bhv + r * ahh[v]);
    float hold = hcur[m * DDIM + c.ncol];
    float hn = z * hold + (1.0f - z) * hh;
    hnext[m * DDIM + c.ncol] = hn;
    out_t[(size_t)m * (TSTEPS * DDIM) + c.ncol] = hn;      // out[b][t][d]
    int lane2 = (m & 15) + c.lane2b;
    afrag_next[((((m >> 4) * KT + c.ktile) * 32 + lane2) << 4) + c.j2] = f2bf(hn);
  }
}

// Device-wide sense barrier: bar[0]=arrive count, bar[1]=generation.
__device__ __forceinline__ void grid_barrier(unsigned* __restrict__ bar, int t) {
  unsigned* count = bar;
  unsigned* gen   = bar + 1;
  __threadfence();                 // make this thread's stores agent-visible
  __syncthreads();                 // all threads of block fenced
  if (threadIdx.x == 0) {
    unsigned arrived = __hip_atomic_fetch_add(count, 1u, __ATOMIC_ACQ_REL,
                                              __HIP_MEMORY_SCOPE_AGENT);
    if (arrived == (unsigned)(NBLK - 1)) {
      __hip_atomic_store(count, 0u, __ATOMIC_RELAXED, __HIP_MEMORY_SCOPE_AGENT);
      __hip_atomic_fetch_add(gen, 1u, __ATOMIC_RELEASE, __HIP_MEMORY_SCOPE_AGENT);
    } else {
      while (__hip_atomic_load(gen, __ATOMIC_ACQUIRE, __HIP_MEMORY_SCOPE_AGENT)
             < (unsigned)(t + 1)) {
        __builtin_amdgcn_s_sleep(2);
      }
    }
  }
  __syncthreads();
  __threadfence();                 // acquire: drop stale near-cache lines
}

// ---------------------------------------------------------------------------
// Persistent GRU: 128 blocks x 8 waves; each wave owns one 16x16 tile of
// new_h per step.
// ---------------------------------------------------------------------------
__global__ __launch_bounds__(256) void gru_persistent_kernel(
    const float* __restrict__ x,
    const unsigned short* __restrict__ wfrag,
    const unsigned short* __restrict__ ufrag,
    const float* __restrict__ bias,
    unsigned short* __restrict__ abuf0, unsigned short* __restrict__ abuf1,
    float* __restrict__ hbuf0, float* __restrict__ hbuf1,
    float* __restrict__ out, unsigned* __restrict__ bar)
{
  const int wid = (blockIdx.x * 256 + threadIdx.x) >> 5;   // 0..1023
  TileCtx c;
  c.lane = threadIdx.x & 31;
  c.nt   = wid >> 4;                 // 0..63 : n-tile within D
  c.mt   = wid & 15;                 // 0..15 : m-tile within batch
  c.half = c.lane >> 4;
  c.bz0  = (c.nt)       * (KT * 32) + c.lane;
  c.br0  = (c.nt + 64)  * (KT * 32) + c.lane;
  c.bh0  = (c.nt + 128) * (KT * 32) + c.lane;
  c.ncol = (c.nt << 4) + (c.lane & 15);
  c.bzv  = bias[c.ncol];
  c.brv  = bias[DDIM + c.ncol];
  c.bhv  = bias[2 * DDIM + c.ncol];
  const int kin = c.ncol & 31;
  c.ktile  = c.ncol >> 5;
  c.lane2b = ((kin >> 3) & 1) << 4;
  c.j2     = (kin & 7) + ((kin >> 4) << 3);

  const v16bf* wv = reinterpret_cast<const v16bf*>(wfrag);
  const v16bf* uv = reinterpret_cast<const v16bf*>(ufrag);

  // ---- step 0 (peeled): inp == 0, state == x ----
  gru_step_body<true>(c, abuf0, x, wv, uv, hbuf1, abuf1, out);
  grid_barrier(bar, 0);

  // ---- steady state: t = 1..127, branch-free inner loop ----
  for (int t = 1; t < TSTEPS; ++t) {
    const unsigned short* afrag = (t & 1) ? abuf1 : abuf0;
    unsigned short* afrag_next  = (t & 1) ? abuf0 : abuf1;
    const float* hcur = (t & 1) ? hbuf1 : hbuf0;
    float* hnext      = (t & 1) ? hbuf0 : hbuf1;
    gru_step_body<false>(c, afrag, hcur, wv, uv, hnext, afrag_next,
                         out + (size_t)t * DDIM);
    grid_barrier(bar, t);
  }
}

// ---------------------------------------------------------------------------
extern "C" void kernel_launch(void* const* d_in, const int* in_sizes, int n_in,
                              void* d_out, int out_size, void* d_ws, size_t ws_size,
                              hipStream_t stream) {
  (void)in_sizes; (void)n_in; (void)out_size; (void)ws_size;
  const float* x = (const float*)d_in[0];   // [256,1024]  initial state
  const float* W = (const float*)d_in[1];   // [1024,3072]
  const float* U = (const float*)d_in[2];   // [1024,3072]
  const float* b = (const float*)d_in[3];   // [3072]
  float* out = (float*)d_out;               // [256,128,1024]

  unsigned short* ws = (unsigned short*)d_ws;
  unsigned short* wfrag  = ws;                       // 6 MB
  unsigned short* ufrag  = wfrag + WFRAG_ELEMS;      // 6 MB
  unsigned short* afrag0 = ufrag + WFRAG_ELEMS;      // 512 KB
  unsigned short* afrag1 = afrag0 + AFRAG_ELEMS;     // 512 KB
  float* h0 = (float*)(afrag1 + AFRAG_ELEMS);        // 1 MB
  float* h1 = h0 + (size_t)NB * DDIM;                // 1 MB
  unsigned* bar = (unsigned*)(h1 + (size_t)NB * DDIM);

  // one-time (per launch) weight conversion to bf16 fragments
  pack_weight_kernel<<<(int)(WFRAG_ELEMS / 256), 256, 0, stream>>>(W, wfrag);
  pack_weight_kernel<<<(int)(WFRAG_ELEMS / 256), 256, 0, stream>>>(U, ufrag);
  pack_a_kernel<<<(int)(AFRAG_ELEMS / 256), 256, 0, stream>>>(x, afrag0);
  init_barrier_kernel<<<1, 32, 0, stream>>>(bar);

  // all 128 autoregressive steps inside one persistent kernel
  gru_persistent_kernel<<<NBLK, 256, 0, stream>>>(x, wfrag, ufrag, b,
                                                  afrag0, afrag1, h0, h1,
                                                  out, bar);
}
